// SASRec_27230092657010
// MI455X (gfx1250) — matrix operand order repaired and checked by reading, not compiled, not run on hardware
//
#include <hip/hip_runtime.h>

// ---------------- problem constants ----------------
#define B_   128
#define L_   512
#define E_   512
#define H_   8
#define HD_  64
#define NB_  2
#define WIN_ 64
#define EPSF 1e-8f

typedef __attribute__((ext_vector_type(16))) __bf16 v16bf;
typedef __attribute__((ext_vector_type(8)))  float  v8f;

// ---------------- CDNA5 async copy (global -> LDS) ----------------
#if __has_builtin(__builtin_amdgcn_global_load_async_to_lds_b128)
#define ASYNC_LDS 1
#else
#define ASYNC_LDS 0
#endif

typedef int v4i __attribute__((ext_vector_type(4)));
typedef __attribute__((address_space(1))) v4i* gv4i_p;   // global int4*
typedef __attribute__((address_space(3))) v4i* lv4i_p;   // LDS int4*

// Global generic addresses are identity-mapped into as(1); LDS generic
// addresses carry the LDS byte offset in their low 32 bits, so these integer
// round-trips produce value-correct segment pointers.
#define TO_GLOBAL(p) ((gv4i_p)(uintptr_t)(p))
#define TO_LDS(p)    ((lv4i_p)(uint32_t)(uintptr_t)(p))

// Stage 32 bytes (16 bf16) from global to LDS.
__device__ __forceinline__ void stage32B(const __bf16* g, __bf16* l) {
#if ASYNC_LDS
    __builtin_amdgcn_global_load_async_to_lds_b128(TO_GLOBAL(g),     TO_LDS(l),     0, 0);
    __builtin_amdgcn_global_load_async_to_lds_b128(TO_GLOBAL(g + 8), TO_LDS(l + 8), 0, 0);
#else
    uint4 x0 = *reinterpret_cast<const uint4*>(g);
    uint4 x1 = *reinterpret_cast<const uint4*>(g + 8);
    *reinterpret_cast<uint4*>(l)     = x0;
    *reinterpret_cast<uint4*>(l + 8) = x1;
#endif
}

__device__ __forceinline__ void wait_async4() {
#if ASYNC_LDS
#if __has_builtin(__builtin_amdgcn_s_wait_asynccnt)
    __builtin_amdgcn_s_wait_asynccnt(4);
#else
    asm volatile("s_wait_asynccnt 0x4" ::: "memory");
#endif
#endif
}
__device__ __forceinline__ void wait_async0() {
#if ASYNC_LDS
#if __has_builtin(__builtin_amdgcn_s_wait_asynccnt)
    __builtin_amdgcn_s_wait_asynccnt(0);
#else
    asm volatile("s_wait_asynccnt 0x0" ::: "memory");
#endif
#endif
}

// ---------------- WMMA fragment helpers ----------------
__device__ __forceinline__ void ld16B(void* dst, const void* src) {
    *reinterpret_cast<uint4*>(dst) = *reinterpret_cast<const uint4*>(src);
}

// A-fragment (16x32 bf16, MxK): lane holds row M=lane%16.
// lanes 0-15: elems[0..15]=K kb+0..15
// lanes16-31: elems[0..15]=K kb+16..31 (contiguous)
__device__ __forceinline__ v16bf make_afrag(const __bf16* rowptr, int lane, int kb) {
    v16bf f;
    int o = (lane < 16) ? 0 : 16;
    ld16B(reinterpret_cast<uint4*>(&f) + 0, rowptr + kb + o);
    ld16B(reinterpret_cast<uint4*>(&f) + 1, rowptr + kb + o + 8);
    return f;
}

// B-fragment (32x16 bf16, KxN): lane holds column N=lane%16.
// lanes 0-15 hold K kb+0..15, lanes 16-31 hold K kb+16..31 (contiguous).
__device__ __forceinline__ v16bf make_bfrag(const __bf16* colptr, int lane, int kb) {
    v16bf f;
    int o = kb + ((lane < 16) ? 0 : 16);
    ld16B(reinterpret_cast<uint4*>(&f) + 0, colptr + o);
    ld16B(reinterpret_cast<uint4*>(&f) + 1, colptr + o + 8);
    return f;
}

__device__ __forceinline__ v8f wmma_bf16(v16bf a, v16bf b, v8f c) {
    return __builtin_amdgcn_wmma_f32_16x16x32_bf16(false, a, false, b, (short)0, c,
                                                   false, false);
}

// ---------------- elementwise kernels ----------------
__global__ void k_f2bf(const float* __restrict__ in, __bf16* __restrict__ out, int n) {
    int i = blockIdx.x * 256 + threadIdx.x;
    if (i < n) out[i] = (__bf16)in[i];
}

// x = in * keep; write fp32 + bf16 copies. (in may alias outf)
__global__ void k_maskmul(const float* in, const unsigned char* __restrict__ mask,
                          float* outf, __bf16* __restrict__ outb) {
    int i = blockIdx.x * 256 + threadIdx.x;        // float4 index
    int row = i / (E_ / 4);
    float k = mask[row] ? 0.f : 1.f;
    float4 v = reinterpret_cast<const float4*>(in)[i];
    float4 o; o.x = v.x * k; o.y = v.y * k; o.z = v.z * k; o.w = v.w * k;
    reinterpret_cast<float4*>(outf)[i] = o;
    union { __bf16 h[4]; unsigned long long u; } pk;
    pk.h[0] = (__bf16)o.x; pk.h[1] = (__bf16)o.y;
    pk.h[2] = (__bf16)o.z; pk.h[3] = (__bf16)o.w;
    reinterpret_cast<unsigned long long*>(outb)[i] = pk.u;
}

// wave-per-row LayerNorm over E=512; optional fp32 / bf16 outputs
__global__ void k_layernorm(const float* __restrict__ in, const float* __restrict__ g,
                            const float* __restrict__ bt, float* __restrict__ outf,
                            __bf16* __restrict__ outb) {
    int lane = threadIdx.x & 31, wid = threadIdx.x >> 5;
    size_t row = (size_t)blockIdx.x * 8 + wid;
    const float* r = in + row * E_;
    float4 v[4];
    float s = 0.f, q = 0.f;
#pragma unroll
    for (int c = 0; c < 4; c++) {
        v[c] = reinterpret_cast<const float4*>(r)[lane + c * 32];
        s += v[c].x + v[c].y + v[c].z + v[c].w;
        q += v[c].x * v[c].x + v[c].y * v[c].y + v[c].z * v[c].z + v[c].w * v[c].w;
    }
#pragma unroll
    for (int m = 1; m < 32; m <<= 1) { s += __shfl_xor(s, m, 32); q += __shfl_xor(q, m, 32); }
    float mean = s * (1.f / E_);
    float var  = q * (1.f / E_) - mean * mean;
    float inv  = rsqrtf(var + EPSF);
#pragma unroll
    for (int c = 0; c < 4; c++) {
        int idx = lane + c * 32;
        float4 gg = reinterpret_cast<const float4*>(g)[idx];
        float4 bb = reinterpret_cast<const float4*>(bt)[idx];
        float4 o;
        o.x = (v[c].x - mean) * inv * gg.x + bb.x;
        o.y = (v[c].y - mean) * inv * gg.y + bb.y;
        o.z = (v[c].z - mean) * inv * gg.z + bb.z;
        o.w = (v[c].w - mean) * inv * gg.w + bb.w;
        if (outf) reinterpret_cast<float4*>(outf + row * E_)[idx] = o;
        if (outb) {
            union { __bf16 h[4]; unsigned long long u; } pk;
            pk.h[0] = (__bf16)o.x; pk.h[1] = (__bf16)o.y;
            pk.h[2] = (__bf16)o.z; pk.h[3] = (__bf16)o.w;
            reinterpret_cast<unsigned long long*>(outb + row * E_)[idx] = pk.u;
        }
    }
}

// v: [B,L,H,HD] -> vT: [B,H,HD,L]
__global__ void k_transpose_v(const __bf16* __restrict__ v, __bf16* __restrict__ vT) {
    size_t i = (size_t)blockIdx.x * 256 + threadIdx.x;   // output index
    int l = (int)(i % L_); size_t r = i / L_;
    int d = (int)(r % HD_); r /= HD_;
    int h = (int)(r % H_);  int b = (int)(r / H_);
    vT[i] = v[(((size_t)b * L_ + l) * H_ + h) * HD_ + d];
}

// ---------------- WMMA GEMM: C = act(A[M,K] @ W[N,K]^T + bias (+resid)) ----------------
// 128x128 block tile, 8 waves (4M x 2N), double-buffered LDS fed by async copies.
__global__ __launch_bounds__(256, 2)
void k_gemm_bf16(const __bf16* __restrict__ A, const __bf16* __restrict__ W,
                 const float* __restrict__ bias, const float* __restrict__ resid,
                 float* __restrict__ Cf, __bf16* __restrict__ Cb,
                 int M, int N, int K, int relu) {
    __shared__ __align__(16) __bf16 As[2][128 * 32];
    __shared__ __align__(16) __bf16 Bs[2][128 * 32];
    int tiles_n = N >> 7;
    int m0 = (blockIdx.x / tiles_n) << 7;
    int n0 = (blockIdx.x % tiles_n) << 7;
    int t = threadIdx.x, lane = t & 31, wid = t >> 5;
    int wm = wid & 3, wn = wid >> 2;          // 4 M-subtiles x 2 N-subtiles
    int lm = lane & 15;

    // staging map: thread t stages 32B of row (t>>1) at k-offset (t&1)*16
    int srow = t >> 1, sko = (t & 1) << 4;
    const __bf16* Ag = A + (size_t)(m0 + srow) * K + sko;
    const __bf16* Wg = W + (size_t)(n0 + srow) * K + sko;
    __bf16* Al0 = &As[0][srow * 32 + sko];
    __bf16* Bl0 = &Bs[0][srow * 32 + sko];
    __bf16* Al1 = &As[1][srow * 32 + sko];
    __bf16* Bl1 = &Bs[1][srow * 32 + sko];

    v8f acc[2][4];
#pragma unroll
    for (int a = 0; a < 2; a++)
#pragma unroll
        for (int b = 0; b < 4; b++)
#pragma unroll
            for (int e = 0; e < 8; e++) acc[a][b][e] = 0.f;

    const int nk = K >> 5;
    // prologue: tile 0 -> buffer 0
    stage32B(Ag, Al0);
    stage32B(Wg, Bl0);

    for (int it = 0; it < nk; ++it) {
        int cur = it & 1;
        if (it + 1 < nk) {                    // issue next tile into other buffer
            const __bf16* an = Ag + (size_t)(it + 1) * 32;
            const __bf16* wn2 = Wg + (size_t)(it + 1) * 32;
            stage32B(an,  cur ? Al0 : Al1);
            stage32B(wn2, cur ? Bl0 : Bl1);
            __builtin_prefetch(an + 64, 0, 0);   // 2 tiles ahead
            __builtin_prefetch(wn2 + 64, 0, 0);
            wait_async4();                    // tile `it` complete (in-order)
        } else {
            wait_async0();
        }
        __syncthreads();

        const __bf16* Asb = &As[cur][0];
        const __bf16* Bsb = &Bs[cur][0];
        v16bf af[2], bf4[4];
#pragma unroll
        for (int mf = 0; mf < 2; mf++)
            af[mf] = make_afrag(Asb + (wm * 32 + mf * 16 + lm) * 32, lane, 0);
#pragma unroll
        for (int nf = 0; nf < 4; nf++)
            bf4[nf] = make_bfrag(Bsb + (wn * 64 + nf * 16 + lm) * 32, lane, 0);
#pragma unroll
        for (int mf = 0; mf < 2; mf++)
#pragma unroll
            for (int nf = 0; nf < 4; nf++)
                acc[mf][nf] = wmma_bf16(af[mf], bf4[nf], acc[mf][nf]);
        __syncthreads();                      // done reading buf `cur`
    }

    // epilogue: C VGPR layout -> rows e / e+8, col = lane%16
#pragma unroll
    for (int mf = 0; mf < 2; mf++)
#pragma unroll
        for (int nf = 0; nf < 4; nf++) {
            int coln = n0 + wn * 64 + nf * 16 + lm;
            float bv = bias ? bias[coln] : 0.f;
#pragma unroll
            for (int e = 0; e < 8; e++) {
                int rowm = m0 + wm * 32 + mf * 16 + ((lane < 16) ? e : e + 8);
                float val = acc[mf][nf][e] + bv;
                if (resid) val += resid[(size_t)rowm * N + coln];
                if (relu)  val = fmaxf(val, 0.f);
                if (Cf) Cf[(size_t)rowm * N + coln] = val;
                if (Cb) Cb[(size_t)rowm * N + coln] = (__bf16)val;
            }
        }
}

// ---------------- sliding-window attention (wave per (b,h,16-query tile)) ----------------
__global__ __launch_bounds__(256, 2)
void k_attn(const __bf16* __restrict__ q, const __bf16* __restrict__ kmat,
            const __bf16* __restrict__ vT, __bf16* __restrict__ ob) {
    __shared__ __align__(16) __bf16 Ps[8][16 * 96];   // per-wave P tile (16x80, padded to 96)
    int lane = threadIdx.x & 31, wid = threadIdx.x >> 5;
    int g = blockIdx.x * 8 + wid;
    int it = g % (L_ / 16); g /= (L_ / 16);
    int h = g % H_; int b = g / H_;
    int i0 = it * 16;
    int jb = i0 - 64;                                 // key window [jb, jb+80)
    int lm = lane & 15;

    // Q fragments (HD=64 -> two K-steps of 32)
    const __bf16* qr = q + ((size_t)b * L_ + (i0 + lm)) * E_ + h * HD_;
    v16bf qf0 = make_afrag(qr, lane, 0);
    v16bf qf1 = make_afrag(qr, lane, 32);

    v8f s[5];
#pragma unroll
    for (int kt = 0; kt < 5; kt++)
#pragma unroll
        for (int e = 0; e < 8; e++) s[kt][e] = 0.f;

#pragma unroll
    for (int kt = 0; kt < 5; kt++) {
        int j = jb + kt * 16 + lm;
        int jc = j < 0 ? 0 : j;
        const __bf16* kr = kmat + ((size_t)b * L_ + jc) * E_ + h * HD_;
        v16bf b0 = make_bfrag(kr, lane, 0);
        v16bf b1 = make_bfrag(kr, lane, 32);
        s[kt] = wmma_bf16(qf0, b0, s[kt]);
        s[kt] = wmma_bf16(qf1, b1, s[kt]);
    }

    // scale + window mask + row softmax (rows split lanes<16 / >=16)
    const float scale = 0.125f;                       // 1/sqrt(64)
    float p[5][8];
#pragma unroll
    for (int e = 0; e < 8; e++) {
        int i = i0 + ((lane < 16) ? e : e + 8);
        float mx = -1e30f;
#pragma unroll
        for (int kt = 0; kt < 5; kt++) {
            int j = jb + kt * 16 + lm;
            int d = i - j;
            float val = (j >= 0 && d >= 0 && d < WIN_) ? s[kt][e] * scale : -1e30f;
            p[kt][e] = val;
            mx = fmaxf(mx, val);
        }
#pragma unroll
        for (int m = 1; m < 16; m <<= 1) mx = fmaxf(mx, __shfl_xor(mx, m, 32));
        float sm = 0.f;
#pragma unroll
        for (int kt = 0; kt < 5; kt++) { float pe = __expf(p[kt][e] - mx); p[kt][e] = pe; sm += pe; }
#pragma unroll
        for (int m = 1; m < 16; m <<= 1) sm += __shfl_xor(sm, m, 32);
        float inv = 1.f / sm;
#pragma unroll
        for (int kt = 0; kt < 5; kt++) p[kt][e] *= inv;
    }

    // stage P to LDS (bf16, row-major 16x96) to re-form A-fragments
    __bf16* ps = &Ps[wid][0];
#pragma unroll
    for (int kt = 0; kt < 5; kt++)
#pragma unroll
        for (int e = 0; e < 8; e++) {
            int mrow = (lane < 16) ? e : e + 8;
            ps[mrow * 96 + kt * 16 + lm] = (__bf16)p[kt][e];
        }
    {   // zero the 16-col K-padding (cols 80..95)
        int cb = 80 + ((lane >> 4) << 3);
#pragma unroll
        for (int c = 0; c < 8; c++) ps[lm * 96 + cb + c] = (__bf16)0.f;
    }

    // O = P(16x96) @ V(96x64): 3 K-steps x 4 N-tiles
    v8f o4[4];
#pragma unroll
    for (int nt = 0; nt < 4; nt++)
#pragma unroll
        for (int e = 0; e < 8; e++) o4[nt][e] = 0.f;

#pragma unroll
    for (int kq = 0; kq < 3; kq++) {
        v16bf pa = make_afrag(ps + lm * 96, lane, kq * 32);
        int jc0 = jb + kq * 32 + ((lane < 16) ? 0 : 16);
        if (jc0 < 0) jc0 = 0;
        if (jc0 > L_ - 16) jc0 = L_ - 16;
#pragma unroll
        for (int nt = 0; nt < 4; nt++) {
            const __bf16* vr = vT + (((size_t)b * H_ + h) * HD_ + nt * 16 + lm) * L_ + jc0;
            v16bf vb;
            ld16B(reinterpret_cast<uint4*>(&vb) + 0, vr);
            ld16B(reinterpret_cast<uint4*>(&vb) + 1, vr + 8);
            o4[nt] = wmma_bf16(pa, vb, o4[nt]);
        }
    }

#pragma unroll
    for (int nt = 0; nt < 4; nt++)
#pragma unroll
        for (int e = 0; e < 8; e++) {
            int i = i0 + ((lane < 16) ? e : e + 8);
            ob[((size_t)b * L_ + i) * E_ + h * HD_ + nt * 16 + lm] = (__bf16)o4[nt][e];
        }
}

// ---------------- host orchestration ----------------
extern "C" void kernel_launch(void* const* d_in, const int* in_sizes, int n_in,
                              void* d_out, int out_size, void* d_ws, size_t ws_size,
                              hipStream_t stream) {
    const unsigned char* mask = (const unsigned char*)d_in[0];
    const float* seqs = (const float*)d_in[1];
    const float* w_in = (const float*)d_in[2];
    const float* b_in = (const float*)d_in[3];
    const float* w_out = (const float*)d_in[4];
    const float* b_out = (const float*)d_in[5];
    const float* ln1g = (const float*)d_in[6];
    const float* ln1b = (const float*)d_in[7];
    const float* ln2g = (const float*)d_in[8];
    const float* ln2b = (const float*)d_in[9];
    const float* c1w = (const float*)d_in[10];
    const float* c1b = (const float*)d_in[11];
    const float* c2w = (const float*)d_in[12];
    const float* c2b = (const float*)d_in[13];
    const float* lnfg = (const float*)d_in[14];
    const float* lnfb = (const float*)d_in[15];

    char* ws = (char*)d_ws;
    const size_t FB = (size_t)B_ * L_ * E_ * sizeof(float);   // 134 MB
    const size_t HB = FB / 2;                                  // bf16 copy

    float*  xf   = (float*)(ws + 0 * FB);
    float*  qinf = (float*)(ws + 1 * FB);
    float*  x2f  = (float*)(ws + 2 * FB);
    char*   bfs  = ws + 3 * FB;
    __bf16* xb   = (__bf16*)(bfs + 0 * HB);
    __bf16* qinb = (__bf16*)(bfs + 1 * HB);
    __bf16* qb   = (__bf16*)(bfs + 2 * HB);
    __bf16* kb   = (__bf16*)(bfs + 3 * HB);
    __bf16* vb   = (__bf16*)(bfs + 4 * HB);
    __bf16* vT   = (__bf16*)(bfs + 5 * HB);
    __bf16* obuf = (__bf16*)(bfs + 6 * HB);
    __bf16* x2b  = (__bf16*)(bfs + 7 * HB);
    __bf16* hb   = (__bf16*)(bfs + 8 * HB);
    __bf16* wqkvb = (__bf16*)(bfs + 9 * HB);
    __bf16* woutb = wqkvb + (size_t)NB_ * 3 * E_ * E_;
    __bf16* c1wb  = woutb + (size_t)NB_ * E_ * E_;
    __bf16* c2wb  = c1wb  + (size_t)NB_ * E_ * E_;

    const int rows  = B_ * L_;                 // 65536
    const int nvec4 = rows * E_ / 4;
    const int nw1 = NB_ * 3 * E_ * E_;
    const int nw2 = NB_ * E_ * E_;

    k_f2bf<<<(nw1 + 255) / 256, 256, 0, stream>>>(w_in,  wqkvb, nw1);
    k_f2bf<<<(nw2 + 255) / 256, 256, 0, stream>>>(w_out, woutb, nw2);
    k_f2bf<<<(nw2 + 255) / 256, 256, 0, stream>>>(c1w,   c1wb,  nw2);
    k_f2bf<<<(nw2 + 255) / 256, 256, 0, stream>>>(c2w,   c2wb,  nw2);

    k_maskmul<<<nvec4 / 256, 256, 0, stream>>>(seqs, mask, xf, xb);

    const int gG = (rows / 128) * (E_ / 128);  // 2048 blocks per GEMM

    for (int blk = 0; blk < NB_; blk++) {
        const __bf16* wq = wqkvb + (size_t)blk * 3 * E_ * E_;
        const __bf16* wk = wq + (size_t)E_ * E_;
        const __bf16* wv = wk + (size_t)E_ * E_;
        const float*  bq = b_in + blk * 3 * E_;

        k_layernorm<<<rows / 8, 256, 0, stream>>>(xf, ln1g + blk * E_, ln1b + blk * E_,
                                                  qinf, qinb);
        // q = LN1(x) @ wq^T ; k,v from raw (masked) x
        k_gemm_bf16<<<gG, 256, 0, stream>>>(qinb, wq, bq,          nullptr, nullptr, qb,
                                            rows, E_, E_, 0);
        k_gemm_bf16<<<gG, 256, 0, stream>>>(xb,   wk, bq + E_,     nullptr, nullptr, kb,
                                            rows, E_, E_, 0);
        k_gemm_bf16<<<gG, 256, 0, stream>>>(xb,   wv, bq + 2 * E_, nullptr, nullptr, vb,
                                            rows, E_, E_, 0);
        k_transpose_v<<<rows * E_ / 256, 256, 0, stream>>>(vb, vT);
        k_attn<<<(B_ * H_ * (L_ / 16)) / 8, 256, 0, stream>>>(qb, kb, vT, obuf);
        // x = q_in + o @ w_out^T + b_out
        k_gemm_bf16<<<gG, 256, 0, stream>>>(obuf, woutb + (size_t)blk * E_ * E_,
                                            b_out + blk * E_, qinf, xf, nullptr,
                                            rows, E_, E_, 0);
        k_layernorm<<<rows / 8, 256, 0, stream>>>(xf, ln2g + blk * E_, ln2b + blk * E_,
                                                  x2f, x2b);
        // h = relu(x2 @ c1^T + c1_b)
        k_gemm_bf16<<<gG, 256, 0, stream>>>(x2b, c1wb + (size_t)blk * E_ * E_,
                                            c1b + blk * E_, nullptr, nullptr, hb,
                                            rows, E_, E_, 1);
        // x = h @ c2^T + c2_b + x2
        k_gemm_bf16<<<gG, 256, 0, stream>>>(hb, c2wb + (size_t)blk * E_ * E_,
                                            c2b + blk * E_, x2f, xf, nullptr,
                                            rows, E_, E_, 0);
        k_maskmul<<<nvec4 / 256, 256, 0, stream>>>(xf, mask, xf, xb);
    }
    k_layernorm<<<rows / 8, 256, 0, stream>>>(xf, lnfg, lnfb, (float*)d_out, nullptr);
}
